// CrossModalContrastiveLoss_80771154968931
// MI455X (gfx1250) — matrix operand order repaired and checked by reading, not compiled, not run on hardware
//
#include <hip/hip_runtime.h>
#include <math.h>

typedef __attribute__((ext_vector_type(16))) __bf16 v16bf;
typedef __attribute__((ext_vector_type(8)))  __bf16 v8bf;
typedef __attribute__((ext_vector_type(8)))  float  v8f;
typedef __attribute__((ext_vector_type(4)))  unsigned int v4u;
typedef __attribute__((ext_vector_type(8)))  int v8i;
typedef __attribute__((ext_vector_type(4)))  int v4i;

#define N_ROWS    8192
#define CDIM      256
#define ROWS_PB   32          // rows per block (2 row-tiles per wave)
#define INV_TEMP  10.0f

// ---------------------------------------------------------------------------
// Kernel 1: (B,C,H,W) -> row-major (N,C), L2-normalized over C, bf16 convert.
// ---------------------------------------------------------------------------
__global__ void normalize_transpose_kernel(const float* __restrict__ rgb,
                                           const float* __restrict__ xf,
                                           __bf16* __restrict__ Abf,
                                           __bf16* __restrict__ Xbf) {
    __shared__ float2 sred[256];
    const int n  = blockIdx.x;
    const int c  = threadIdx.x;
    const int b  = n >> 10;
    const int hw = n & 1023;
    const size_t idx = ((size_t)(b * CDIM + c) << 10) + (size_t)hw;

    const float vr = rgb[idx];
    const float vx = xf[idx];

    sred[c] = make_float2(vr * vr, vx * vx);
    __syncthreads();
    #pragma unroll
    for (int s = 128; s > 0; s >>= 1) {
        if (c < s) {
            sred[c].x += sred[c + s].x;
            sred[c].y += sred[c + s].y;
        }
        __syncthreads();
    }
    const float ir = 1.0f / fmaxf(sqrtf(sred[0].x), 1e-12f);
    const float ix = 1.0f / fmaxf(sqrtf(sred[0].y), 1e-12f);

    Abf[(size_t)n * CDIM + c] = (__bf16)(vr * ir);
    Xbf[(size_t)n * CDIM + c] = (__bf16)(vx * ix);
}

// ---------------------------------------------------------------------------
// Kernel 2: fused sim GEMM + row-wise sum(exp) + masked sim sum.
// Block = 8 waves, 32-row band staged to LDS by the Tensor Data Mover.
// Each wave: 2 row-tiles (B fragment reused by 2 WMMAs), streams every 8th 16-col tile of 8192
// columns; next tile's B loads issued before the exp epilogue for overlap.
// ---------------------------------------------------------------------------
__device__ __forceinline__ void load_btile(const __bf16* __restrict__ X,
                                           int col, int ko, v16bf b[8]) {
    const __bf16* xrow = X + (size_t)col * CDIM + ko;
    #pragma unroll
    for (int kb = 0; kb < 8; ++kb)
        b[kb] = __builtin_shufflevector(*(const v8bf*)(xrow + kb * 32), *(const v8bf*)(xrow + kb * 32 + 16), 0, 1, 2, 3, 4, 5, 6, 7, 8, 9, 10, 11, 12, 13, 14, 15);
}

__global__ void __launch_bounds__(256, 1)
sim_lse_kernel(const __bf16* __restrict__ A,
               const __bf16* __restrict__ X,
               float* __restrict__ partials) {
    __shared__ __bf16 sA[ROWS_PB][CDIM];        // 16 KB A band (TDM-staged)
    __shared__ float  s_e[8][ROWS_PB];
    __shared__ float  s_p[8][ROWS_PB];

    const int tid   = threadIdx.x;
    const int lane  = tid & 31;
    const int wave  = tid >> 5;
    const int M0    = blockIdx.x * ROWS_PB;
    const int batch = blockIdx.x >> 5;          // M0 / 1024 (band within one batch)

    // ---- TDM: stage contiguous 32x256 bf16 tile (16 KB) into LDS ---------
    if (wave == 0) {
        const unsigned long long ga =
            (unsigned long long)(const void*)(A + (size_t)M0 * CDIM);
        const unsigned lds_off = (unsigned)(size_t)(void*)&sA[0][0];
        v4u g0;
        g0[0] = 1u;                                      // count=1 (valid user D#)
        g0[1] = lds_off;                                 // lds_addr (bytes)
        g0[2] = (unsigned)ga;                            // global_addr[31:0]
        g0[3] = (unsigned)((ga >> 32) & 0x01FFFFFFu)     // global_addr[56:32]
              | (2u << 30);                              // type=2 ("image")
        v8i g1;
        g1[0] = 0x00010000;                              // wg_mask=0, data_size=2B
        g1[1] = (int)(CDIM << 16);                       // tensor_dim0 = 256
        g1[2] = (int)(N_ROWS << 16);                     // tensor_dim1 = 8192
        g1[3] = (int)(CDIM << 16);                       // tile_dim0 = 256
        g1[4] = ROWS_PB;                                 // tile_dim1 = 32
        g1[5] = CDIM;                                    // tensor_dim0_stride = 256
        g1[6] = 0;
        g1[7] = 0;
        v4i zz = {0, 0, 0, 0};
#if __clang_major__ >= 23
        v8i z8 = {0, 0, 0, 0, 0, 0, 0, 0};
        __builtin_amdgcn_tensor_load_to_lds(g0, g1, zz, zz, z8, 0);
#else
        __builtin_amdgcn_tensor_load_to_lds(g0, g1, zz, zz, 0);
#endif
        __builtin_amdgcn_s_wait_tensorcnt(0);
    }
    __syncthreads();

    // ---- build A fragments for 2 row-tiles from LDS ----------------------
    // 16-bit A 16x32 layout: lanes 0-15 K={0..7,16..23}, lanes 16-31 K={8..15,24..31}
    const int mlocal = lane & 15;
    const int hiK    = lane >> 4;
    const int lo     = hiK ? 8 : 0;
    v16bf af0[8], af1[8];
    #pragma unroll
    for (int kb = 0; kb < 8; ++kb) {
        const int kbase = kb * 32;
        v8bf c0 = *(const v8bf*)&sA[mlocal][kbase + lo];
        v8bf c1 = *(const v8bf*)&sA[mlocal][kbase + 16 + lo];
        v8bf d0 = *(const v8bf*)&sA[16 + mlocal][kbase + lo];
        v8bf d1 = *(const v8bf*)&sA[16 + mlocal][kbase + 16 + lo];
        v16bf a, b;
        #pragma unroll
        for (int e = 0; e < 8; ++e) {
            a[e] = c0[e]; a[8 + e] = c1[e];
            b[e] = d0[e]; b[8 + e] = d1[e];
        }
        af0[kb] = a; af1[kb] = b;
    }

    float e0s[8], e1s[8], p0s[8], p1s[8];
    #pragma unroll
    for (int v = 0; v < 8; ++v) { e0s[v] = e1s[v] = p0s[v] = p1s[v] = 0.0f; }

    // positive iff column % 8 == batch; tile cols 16-aligned -> lane predicate
    const bool inmask = ((lane & 7) == batch);
    const int  ncol   = lane & 15;
    const int  ko     = hiK ? 8 : 0;    // B 32x16: lanes 0-15 K={0..7,16..23}, 16-31 K={8..15,24..31}

    v16bf bfr[8];
    int tc0 = wave * 16;
    load_btile(X, tc0 + ncol, ko, bfr);

    for (int tc = tc0; tc < N_ROWS; tc += 128) {
        v8f a0 = {}, a1 = {};
        #pragma unroll
        for (int kb = 0; kb < 8; ++kb) {
            a0 = __builtin_amdgcn_wmma_f32_16x16x32_bf16(
                     false, af0[kb], false, bfr[kb], (short)0, a0, false, false);
            a1 = __builtin_amdgcn_wmma_f32_16x16x32_bf16(
                     false, af1[kb], false, bfr[kb], (short)0, a1, false, false);
        }
        // issue next tile's B loads before the transcendental epilogue
        if (tc + 128 < N_ROWS) load_btile(X, tc + 128 + ncol, ko, bfr);

        #pragma unroll
        for (int v = 0; v < 8; ++v) {
            const float s0 = a0[v] * INV_TEMP;   // |sim| <= 10 -> exp safe in f32
            const float s1 = a1[v] * INV_TEMP;
            e0s[v] += __expf(s0);
            e1s[v] += __expf(s1);
            if (inmask) { p0s[v] += s0; p1s[v] += s1; }
        }
    }

    // ---- deterministic reduction -----------------------------------------
    // C/D layout: VGPR v, lane L -> row (v + 8*(L>=16)), col (L&15).
    #pragma unroll
    for (int v = 0; v < 8; ++v) {
        float e0 = e0s[v], p0 = p0s[v], e1 = e1s[v], p1 = p1s[v];
        #pragma unroll
        for (int m = 1; m <= 8; m <<= 1) {
            e0 += __shfl_xor(e0, m, 32); p0 += __shfl_xor(p0, m, 32);
            e1 += __shfl_xor(e1, m, 32); p1 += __shfl_xor(p1, m, 32);
        }
        if ((lane & 15) == 0) {
            const int r = v + (hiK ? 8 : 0);
            s_e[wave][r]      = e0; s_p[wave][r]      = p0;
            s_e[wave][16 + r] = e1; s_p[wave][16 + r] = p1;
        }
    }
    __syncthreads();

    if (tid < 32) {                              // whole wave 0 active
        float e = 0.0f, p = 0.0f;
        #pragma unroll
        for (int w = 0; w < 8; ++w) { e += s_e[w][tid]; p += s_p[w][tid]; }
        float part = p - 1024.0f * logf(e);      // 1024 positives per row
        #pragma unroll
        for (int m = 1; m <= 16; m <<= 1) part += __shfl_xor(part, m, 32);
        if (tid == 0) partials[blockIdx.x] = part;
    }
}

// ---------------------------------------------------------------------------
// Kernel 3: deterministic tree reduction of 256 block partials -> loss.
// ---------------------------------------------------------------------------
__global__ void finalize_kernel(const float* __restrict__ partials,
                                float* __restrict__ out) {
    __shared__ float sred[256];
    const int t = threadIdx.x;
    sred[t] = partials[t];
    __syncthreads();
    #pragma unroll
    for (int s = 128; s > 0; s >>= 1) {
        if (t < s) sred[t] += sred[t + s];
        __syncthreads();
    }
    if (t == 0) {
        const float mask_sum = 8388608.0f;       // 8192 rows * 1024 positives
        out[0] = -sred[0] / (mask_sum + 1e-8f);  // * WEIGHT (1.0)
    }
}

// ---------------------------------------------------------------------------
extern "C" void kernel_launch(void* const* d_in, const int* in_sizes, int n_in,
                              void* d_out, int out_size, void* d_ws, size_t ws_size,
                              hipStream_t stream) {
    const float* rgb = (const float*)d_in[0];
    const float* xf  = (const float*)d_in[1];

    __bf16* Abf = (__bf16*)d_ws;                            // 4 MB
    __bf16* Xbf = Abf + (size_t)N_ROWS * CDIM;              // 4 MB
    float*  partials = (float*)((char*)d_ws +
                       2 * (size_t)N_ROWS * CDIM * sizeof(__bf16)); // 1 KB

    normalize_transpose_kernel<<<N_ROWS, 256, 0, stream>>>(rgb, xf, Abf, Xbf);
    sim_lse_kernel<<<N_ROWS / ROWS_PB, 256, 0, stream>>>(Abf, Xbf, partials);
    finalize_kernel<<<1, 256, 0, stream>>>(partials, (float*)d_out);
}
